// MultiHeadAttention_51024211477011
// MI455X (gfx1250) — matrix operand order repaired and checked by reading, not compiled, not run on hardware
//
#include <hip/hip_runtime.h>
#include <hip/hip_bf16.h>

typedef __attribute__((ext_vector_type(16))) _Float16 v16h;
typedef __attribute__((ext_vector_type(8)))  _Float16 h8;
typedef __attribute__((ext_vector_type(8)))  float    v8f;
typedef __attribute__((ext_vector_type(4)))  int      i32x4;

#define NB   4
#define SEQ  2048
#define DM   512
#define NH   8
#define HD   64

#if defined(__AMDGCN__) && __has_builtin(__builtin_amdgcn_global_load_async_to_lds_b128) && __has_builtin(__builtin_amdgcn_s_wait_asynccnt)
#define HAVE_ASYNC 1
#else
#define HAVE_ASYNC 0
#endif

#if HAVE_ASYNC
typedef __attribute__((address_space(1))) i32x4* g128p;
typedef __attribute__((address_space(3))) i32x4* l128p;
__device__ __forceinline__ void async_b128(const void* g, void* l) {
    __builtin_amdgcn_global_load_async_to_lds_b128((g128p)g, (l128p)l, 0, 0);
}
#endif

__device__ __forceinline__ v8f wmma_f16(v16h a, v16h b, v8f c) {
    return __builtin_amdgcn_wmma_f32_16x16x32_f16(
        /*neg_a=*/false, a, /*neg_b=*/false, b,
        /*c_mod=*/(short)0, c, /*reuse_a=*/false, /*reuse_b=*/false);
}

// ---------------------------------------------------------------------------
// Projection: out = x @ W^T + bias, repacked to f16.
// One wave computes a 16x64 output tile (one full head's columns) via
// 64 x wmma_f32_16x16x32_f16; the A-fragment is shared by 4 column subtiles.
// transposed==0 : out layout [B, H, S, HD]   (Q, K)
// transposed==1 : out layout [B, H, HD, S]   (V -> contiguous P@V B-frags)
// ---------------------------------------------------------------------------
__global__ __launch_bounds__(128) void proj_kernel(
    const float* __restrict__ x,      // [B*S, DM]
    const float* __restrict__ W,      // [DM, DM], out = x @ W^T
    const float* __restrict__ bias,   // [DM]
    _Float16* __restrict__ out,
    int transposed)
{
    const int tid  = threadIdx.x;
    const int lane = tid & 31;
    const int wv   = tid >> 5;
    const int tile = blockIdx.x * 4 + wv;     // 4096 tiles (512 rowb x 8 colb)
    const int colb = tile & 7;                // head index (64 cols per head)
    const int rowb = tile >> 3;               // 512 row blocks
    const int row0 = rowb * 16;
    const int col0 = colb * 64;

    const int nlane = lane & 15;
    const int moff  = (lane & 16) ? 8 : 0;    // C/D row offset
    const int aoff  = (lane & 16) ? 16 : 0;   // A-frag K offset (0/16)
    const int boff  = (lane & 16) ? 16 : 0;   // B-frag K offset (0/16)

    const float* arow  = x + (size_t)(row0 + nlane) * DM;
    const float* brow0 = W + (size_t)(col0 +  0 + nlane) * DM;
    const float* brow1 = W + (size_t)(col0 + 16 + nlane) * DM;
    const float* brow2 = W + (size_t)(col0 + 32 + nlane) * DM;
    const float* brow3 = W + (size_t)(col0 + 48 + nlane) * DM;

    v8f c0 = {}, c1 = {}, c2 = {}, c3 = {};
    #pragma unroll 2
    for (int k0 = 0; k0 < DM; k0 += 32) {
        v16h a;
        const float* pa = arow + k0 + aoff;
        #pragma unroll
        for (int i = 0; i < 8; ++i) {
            a[i]     = (_Float16)pa[i];
            a[8 + i] = (_Float16)pa[8 + i];
        }
        v16h b0, b1, b2, b3;
        #pragma unroll
        for (int i = 0; i < 16; ++i) {
            b0[i] = (_Float16)brow0[k0 + boff + i];
            b1[i] = (_Float16)brow1[k0 + boff + i];
            b2[i] = (_Float16)brow2[k0 + boff + i];
            b3[i] = (_Float16)brow3[k0 + boff + i];
        }
        c0 = wmma_f16(a, b0, c0);
        c1 = wmma_f16(a, b1, c1);
        c2 = wmma_f16(a, b2, c2);
        c3 = wmma_f16(a, b3, c3);
    }

    const float bv0 = bias[col0 +  0 + nlane];
    const float bv1 = bias[col0 + 16 + nlane];
    const float bv2 = bias[col0 + 32 + nlane];
    const float bv3 = bias[col0 + 48 + nlane];
    #pragma unroll
    for (int r = 0; r < 8; ++r) {
        const int R  = row0 + r + moff;       // global row = b*SEQ + s
        const int bb = R >> 11;
        const int s  = R & (SEQ - 1);
        const float v0 = c0[r] + bv0, v1 = c1[r] + bv1;
        const float v2 = c2[r] + bv2, v3 = c3[r] + bv3;
        if (!transposed) {
            _Float16* p = out + (((size_t)bb * NH + colb) * SEQ + s) * HD + nlane;
            p[ 0] = (_Float16)v0; p[16] = (_Float16)v1;
            p[32] = (_Float16)v2; p[48] = (_Float16)v3;
        } else {
            _Float16* p = out + (((size_t)bb * NH + colb) * HD + nlane) * SEQ + s;
            p[ 0 * SEQ] = (_Float16)v0; p[16 * SEQ] = (_Float16)v1;
            p[32 * SEQ] = (_Float16)v2; p[48 * SEQ] = (_Float16)v3;
        }
    }
}

// ---------------------------------------------------------------------------
// Stage one 32-key K tile (32x64, contiguous 4KB) and V^T tile (64x32) into
// per-wave LDS. Async path uses global_load_async_to_lds_b128 (ASYNCcnt);
// fallback is a synchronous global->VGPR->LDS copy.
// ---------------------------------------------------------------------------
__device__ __forceinline__ void stage_kv(const _Float16* __restrict__ Kp,
                                         const _Float16* __restrict__ Vp,
                                         int kb, _Float16* kd, _Float16* vd,
                                         int lane)
{
    const _Float16* ks = Kp + (size_t)kb * HD;     // 4KB contiguous
    const int vrow = lane >> 2;                    // 8 rows per pass
    const int vcol = (lane & 3) * 8;               // 16B within 64B row chunk
#if HAVE_ASYNC
    #pragma unroll
    for (int i = 0; i < 8; ++i)
        async_b128(ks + i * 256 + lane * 8, kd + i * 256 + lane * 8);
    #pragma unroll
    for (int i = 0; i < 8; ++i) {
        const int row = i * 8 + vrow;
        async_b128(Vp + (size_t)row * SEQ + kb + vcol, vd + row * 32 + vcol);
    }
#else
    #pragma unroll
    for (int i = 0; i < 8; ++i)
        *(h8*)(kd + i * 256 + lane * 8) = *(const h8*)(ks + i * 256 + lane * 8);
    #pragma unroll
    for (int i = 0; i < 8; ++i) {
        const int row = i * 8 + vrow;
        *(h8*)(vd + row * 32 + vcol) = *(const h8*)(Vp + (size_t)row * SEQ + kb + vcol);
    }
#endif
}

// ---------------------------------------------------------------------------
// Causal flash attention. One wave owns a 16-row query block of one (b,h).
// Double-buffered async K/V staging into LDS overlaps HBM/L2 traffic with
// the 8 wmmas + online-softmax VALU work per 32-key tile.
// ---------------------------------------------------------------------------
__global__ __launch_bounds__(128) void attn_kernel(
    const _Float16* __restrict__ Qh,   // [B,H,S,HD]
    const _Float16* __restrict__ Kh,   // [B,H,S,HD]
    const _Float16* __restrict__ Vt,   // [B,H,HD,S]
    float* __restrict__ out)           // [B,S,DM]
{
    __shared__ _Float16 kbuf[4][2][32 * HD];   // 32KB
    __shared__ _Float16 vbuf[4][2][HD * 32];   // 32KB
    __shared__ _Float16 pbuf[4][16 * 32];      // 4KB

    const int tid  = threadIdx.x;
    const int lane = tid & 31;
    const int wv   = tid >> 5;
    const int w    = blockIdx.x * 4 + wv;      // 4096 waves
    const int qb   = w & 127;
    const int bh   = w >> 7;
    const int q_base = qb << 4;

    const int nlane = lane & 15;
    const int moff  = (lane & 16) ? 8 : 0;
    const int off8  = moff;
    const int off16 = (lane & 16) ? 16 : 0;

    const _Float16* Qp = Qh + (size_t)bh * SEQ * HD;
    const _Float16* Kp = Kh + (size_t)bh * SEQ * HD;
    const _Float16* Vp = Vt + (size_t)bh * HD * SEQ;
    _Float16* pl = pbuf[wv];

    // Q A-fragments for d = 0..31 and 32..63
    v16h qa0, qa1;
    {
        const _Float16* qr = Qp + (size_t)(q_base + nlane) * HD;
        #pragma unroll
        for (int i = 0; i < 8; ++i) {
            qa0[i]     = qr[off16 + i];
            qa0[8 + i] = qr[off16 + 8 + i];
            qa1[i]     = qr[32 + off16 + i];
            qa1[8 + i] = qr[32 + off16 + 8 + i];
        }
    }

    float m[8], l[8];
    v8f o0 = {}, o1 = {}, o2 = {}, o3 = {};
    #pragma unroll
    for (int r = 0; r < 8; ++r) { m[r] = -1e30f; l[r] = 0.0f; }

    const float rs = 0.125f;                   // 1/sqrt(HD)
    const int nblk = (q_base + 15) / 32 + 1;   // causal: tiles with key<=q

    stage_kv(Kp, Vp, 0, kbuf[wv][0], vbuf[wv][0], lane);

    for (int t = 0; t < nblk; ++t) {
        const int kb  = t * 32;
        const int cur = t & 1;
        const bool more = (t + 1) < nblk;
        if (more)
            stage_kv(Kp, Vp, kb + 32, kbuf[wv][cur ^ 1], vbuf[wv][cur ^ 1], lane);
#if HAVE_ASYNC
        if (more) __builtin_amdgcn_s_wait_asynccnt(16);  // current tile done
        else      __builtin_amdgcn_s_wait_asynccnt(0);
        asm volatile("" ::: "memory");
#else
        asm volatile("s_wait_dscnt 0" ::: "memory");
#endif
        const _Float16* kt = kbuf[wv][cur];
        const _Float16* vt = vbuf[wv][cur];

        // ---- scores: S[16x32] = Q(16x64) @ K^T(64x32) ----
        v8f s0 = {}, s1 = {};
        {
            const _Float16* kr0 = kt + (size_t)nlane * HD + off16;
            const _Float16* kr1 = kt + (size_t)(16 + nlane) * HD + off16;
            v16h b00, b01, b10, b11;
            #pragma unroll
            for (int i = 0; i < 16; ++i) {
                b00[i] = kr0[i]; b01[i] = kr0[32 + i];
                b10[i] = kr1[i]; b11[i] = kr1[32 + i];
            }
            s0 = wmma_f16(qa0, b00, s0);
            s0 = wmma_f16(qa1, b01, s0);
            s1 = wmma_f16(qa0, b10, s1);
            s1 = wmma_f16(qa1, b11, s1);
        }

        // ---- causal mask + online softmax ----
        const int key0 = kb + nlane;
        const int key1 = key0 + 16;
        #pragma unroll
        for (int r = 0; r < 8; ++r) {
            const int q = q_base + r + moff;
            float a0 = (key0 <= q) ? s0[r] : -10000.0f;
            float a1 = (key1 <= q) ? s1[r] : -10000.0f;
            float mx = fmaxf(a0, a1);
            mx = fmaxf(mx, __shfl_xor(mx, 1, 32));
            mx = fmaxf(mx, __shfl_xor(mx, 2, 32));
            mx = fmaxf(mx, __shfl_xor(mx, 4, 32));
            mx = fmaxf(mx, __shfl_xor(mx, 8, 32));
            const float mn = fmaxf(m[r], mx);
            const float al = __expf((m[r] - mn) * rs);
            const float p0 = __expf((a0 - mn) * rs);
            const float p1 = __expf((a1 - mn) * rs);
            m[r] = mn;
            l[r] = l[r] * al + p0 + p1;        // per-lane partial; reduced later
            o0[r] *= al; o1[r] *= al; o2[r] *= al; o3[r] *= al;
            const int rr = r + moff;
            pl[rr * 32 + nlane]      = (_Float16)p0;
            pl[rr * 32 + 16 + nlane] = (_Float16)p1;
        }
        // wave-private LDS transpose round-trip (C-layout -> A-layout)
        asm volatile("s_wait_dscnt 0" ::: "memory");

        v16h pa;
        {
            const _Float16* pp = pl + nlane * 32 + off16;
            #pragma unroll
            for (int i = 0; i < 8; ++i) { pa[i] = pp[i]; pa[8 + i] = pp[8 + i]; }
        }

        // ---- O += P(16x32) @ V(32x64) from LDS V^T tile ----
        {
            const _Float16* vr0 = vt + (size_t)( 0 + nlane) * 32 + off16;
            const _Float16* vr1 = vt + (size_t)(16 + nlane) * 32 + off16;
            const _Float16* vr2 = vt + (size_t)(32 + nlane) * 32 + off16;
            const _Float16* vr3 = vt + (size_t)(48 + nlane) * 32 + off16;
            v16h b0, b1, b2, b3;
            #pragma unroll
            for (int i = 0; i < 16; ++i) {
                b0[i] = vr0[i]; b1[i] = vr1[i]; b2[i] = vr2[i]; b3[i] = vr3[i];
            }
            o0 = wmma_f16(pa, b0, o0);
            o1 = wmma_f16(pa, b1, o1);
            o2 = wmma_f16(pa, b2, o2);
            o3 = wmma_f16(pa, b3, o3);
        }
    }

    // ---- finalize: reduce l across the 16-lane half, normalize, store ----
    float inv[8];
    #pragma unroll
    for (int r = 0; r < 8; ++r) {
        float lr = l[r];
        lr += __shfl_xor(lr, 1, 32);
        lr += __shfl_xor(lr, 2, 32);
        lr += __shfl_xor(lr, 4, 32);
        lr += __shfl_xor(lr, 8, 32);
        inv[r] = 1.0f / lr;
    }
    const int b = bh >> 3;
    const int h = bh & 7;
    #pragma unroll
    for (int r = 0; r < 8; ++r) {
        const int s = q_base + r + moff;
        float* orow = out + ((size_t)b * SEQ + s) * DM + h * HD;
        orow[ 0 + nlane] = o0[r] * inv[r];
        orow[16 + nlane] = o1[r] * inv[r];
        orow[32 + nlane] = o2[r] * inv[r];
        orow[48 + nlane] = o3[r] * inv[r];
    }
}

extern "C" void kernel_launch(void* const* d_in, const int* in_sizes, int n_in,
                              void* d_out, int out_size, void* d_ws, size_t ws_size,
                              hipStream_t stream) {
    (void)in_sizes; (void)n_in; (void)out_size; (void)ws_size;
    const float* q  = (const float*)d_in[0];
    const float* k  = (const float*)d_in[1];
    const float* v  = (const float*)d_in[2];
    const float* Wq = (const float*)d_in[3];
    const float* bq = (const float*)d_in[4];
    const float* Wk = (const float*)d_in[5];
    const float* bk = (const float*)d_in[6];
    const float* Wv = (const float*)d_in[7];
    const float* bv = (const float*)d_in[8];
    float* out = (float*)d_out;

    const size_t HALF_PER = (size_t)NB * NH * SEQ * HD;   // 4M halves each
    _Float16* Qh = (_Float16*)d_ws;
    _Float16* Kh = Qh + HALF_PER;
    _Float16* Vt = Kh + HALF_PER;

    dim3 blk(128);
    // projections: 4096 16x64 tiles / 4 waves per block
    proj_kernel<<<1024, blk, 0, stream>>>(q, Wq, bq, Qh, 0);
    proj_kernel<<<1024, blk, 0, stream>>>(k, Wk, bk, Kh, 0);
    proj_kernel<<<1024, blk, 0, stream>>>(v, Wv, bv, Vt, 1);
    // attention: 4096 query-block waves / 4 waves per block
    attn_kernel<<<1024, blk, 0, stream>>>(Qh, Kh, Vt, out);
}
